// Encoder_40209483825845
// MI455X (gfx1250) — matrix operand order repaired and checked by reading, not compiled, not run on hardware
//
#include <hip/hip_runtime.h>
#include <hip/hip_bf16.h>
#include <stdint.h>

// ---------------------------------------------------------------------------
// Types
// ---------------------------------------------------------------------------
typedef __bf16 bf16;
typedef __attribute__((ext_vector_type(16))) __bf16 v16bf;
typedef __attribute__((ext_vector_type(8)))  __bf16 v8bf;
typedef __attribute__((ext_vector_type(8)))  float  v8f;
typedef __attribute__((ext_vector_type(4)))  unsigned int u32x4;
typedef __attribute__((ext_vector_type(4)))  int i32x4;
typedef __attribute__((ext_vector_type(8)))  int i32x8;

#define T_LEN 16384
#define FEAT  512
#define H1    1024
#define G1    4096   // 4*H1 gate rows, layer 1
#define H2    512
#define G2    2048   // 4*H2 gate rows, layer 2
#define NWG   64     // persistent workgroups for the scans (must co-reside)

// ---------------------------------------------------------------------------
// Helpers
// ---------------------------------------------------------------------------
__device__ inline bf16 f2b(float f) {
  union { float f; unsigned int u; } v; v.f = f;
  unsigned int u = v.u;
  unsigned int r = (u + 0x7FFFu + ((u >> 16) & 1u)) >> 16;  // RNE
  union { unsigned short s; bf16 b; } o; o.s = (unsigned short)r;
  return o.b;
}
__device__ inline float b2f(bf16 b) {
  union { unsigned short s; bf16 b; } i; i.b = b;
  union { unsigned int u; float f; } o; o.u = ((unsigned int)i.s) << 16;
  return o.f;
}
__device__ inline float sigf(float x) { return 1.0f / (1.0f + __expf(-x)); }

// Grid-wide barrier: monotonically increasing counter, one arrival per WG.
// Counter is zeroed by a kernel at the start of every launch -> deterministic.
__device__ inline void gsync(unsigned int* cnt, unsigned int* bc) {
  __syncthreads();
  ++(*bc);
  if (threadIdx.x == 0) {
    __threadfence();
    atomicAdd(cnt, 1u);
    const unsigned int tgt = (*bc) * (unsigned int)NWG;
    while (__hip_atomic_load(cnt, __ATOMIC_ACQUIRE, __HIP_MEMORY_SCOPE_AGENT) < tgt)
      __builtin_amdgcn_s_sleep(1);
  }
  __syncthreads();
  __threadfence();
}

// ---------------------------------------------------------------------------
// Tensor Data Mover: load a [tile_d1 x tile_d0] bf16 tile (row stride
// `stride_elems` elements in global) into LDS, packed contiguously.
// Descriptor layout per CDNA5 ISA §8 (2D tensor, groups 2/3 zero).
// ---------------------------------------------------------------------------
#if __has_builtin(__builtin_amdgcn_tensor_load_to_lds)
#define USE_TDM 1
__device__ inline void tdm_load_tile(unsigned int lds_addr, const void* gptr,
                                     unsigned int stride_elems,
                                     unsigned int tile_d0, unsigned int tile_d1)
{
  unsigned long long ga = (unsigned long long)(uintptr_t)gptr;
  u32x4 g0;
  g0[0] = 1u;                                                   // count=1, user D#
  g0[1] = lds_addr;                                             // lds_addr (bytes)
  g0[2] = (unsigned int)(ga & 0xFFFFFFFFull);                   // global_addr[31:0]
  g0[3] = (unsigned int)((ga >> 32) & 0x01FFFFFFull) | (2u << 30); // [56:32] | type=2
  i32x8 g1;
  g1[0] = (int)(1u << 16);                                      // data_size=1 (2B), wg_mask=0
  g1[1] = (int)((tile_d0 & 0xFFFFu) << 16);                     // tensor_dim0 lo16
  g1[2] = (int)(((tile_d0 >> 16) & 0xFFFFu) | ((tile_d1 & 0xFFFFu) << 16)); // dim0 hi / dim1 lo
  g1[3] = (int)(((tile_d1 >> 16) & 0xFFFFu) | ((tile_d0 & 0xFFFFu) << 16)); // dim1 hi / tile_dim0
  g1[4] = (int)(tile_d1 & 0xFFFFu);                             // tile_dim1 (tile_dim2=0)
  g1[5] = (int)stride_elems;                                    // tensor_dim0_stride lo32
  g1[6] = 0;
  g1[7] = 0;
  i32x4 z4 = {0, 0, 0, 0};
#if __clang_major__ >= 23
  i32x8 z8 = {0, 0, 0, 0, 0, 0, 0, 0};
  __builtin_amdgcn_tensor_load_to_lds(g0, g1, z4, z4, z8, 0);
#else
  __builtin_amdgcn_tensor_load_to_lds(g0, g1, z4, z4, 0);
#endif
}
#endif

// ---------------------------------------------------------------------------
// WMMA GEMM: C[M,N] = A[M,K](bf16) * W[N,K]^T(bf16) + (bia[n]+bib[n])
// Workgroup tile 128x128, K-step 32, 8 waves: wave w owns M-strip w*16..+16.
// Double-buffered LDS tiles, TDM prefetch of stage i+1 overlapped with the
// WMMA compute of stage i (TENSORcnt<=2 drain). B fragments move through a
// rotating 2-deep software pipeline: the ds_load_b128 pair for fragment s+1
// is issued before the WMMA that consumes fragment s, so the matrix pipe
// only needs DScnt<=2 instead of a full LDS drain per WMMA.
// Fragment layouts per CDNA5 ISA §7.12.2 (16-bit A 16x32, B 32x16).
// ---------------------------------------------------------------------------
__global__ void __launch_bounds__(256) gemm_bf16_wmma(
    const bf16* __restrict__ A, const bf16* __restrict__ W,
    const float* __restrict__ bia, const float* __restrict__ bib,
    float* __restrict__ C, int N, int K)
{
  __shared__ bf16 As[2 * 128 * 32];
  __shared__ bf16 Ws[2 * 128 * 32];
  const int tid  = threadIdx.x;
  const int wave = tid >> 5;
  const int lane = tid & 31;
  const int r    = lane & 15;
  const int hh   = lane >> 4;
  const int m0   = blockIdx.y * 128;
  const int n0   = blockIdx.x * 128;

  v8f zf = {};
  v8f acc[8];
#pragma unroll
  for (int s = 0; s < 8; ++s) acc[s] = zf;

#if defined(USE_TDM)
  const unsigned int ldsA = (unsigned int)(size_t)As;
  const unsigned int ldsW = (unsigned int)(size_t)Ws;
  // Prologue: fill buffer 0 with the first k-stage.
  if (wave == 0) {
    tdm_load_tile(ldsA, A + (size_t)m0 * K, (unsigned)K, 32, 128);
    tdm_load_tile(ldsW, W + (size_t)n0 * K, (unsigned)K, 32, 128);
  }
#endif

  int ib = 0;
  for (int kk = 0; kk < K; kk += 32, ib ^= 1) {
    __syncthreads();   // all waves done reading buffer ib^1 -> safe to refill
#if defined(USE_TDM)
    if (wave == 0) {
      const int kn = kk + 32;
      if (kn < K) {
        // Prefetch stage i+1 into the other buffer, then drain to <=2 so the
        // current stage's two TDM ops (in-order) are complete.
        tdm_load_tile(ldsA + (unsigned)((ib ^ 1) * 128 * 32 * 2),
                      A + (size_t)m0 * K + kn, (unsigned)K, 32, 128);
        tdm_load_tile(ldsW + (unsigned)((ib ^ 1) * 128 * 32 * 2),
                      W + (size_t)n0 * K + kn, (unsigned)K, 32, 128);
        __builtin_amdgcn_s_wait_tensorcnt(2);
      } else {
        __builtin_amdgcn_s_wait_tensorcnt(0);
      }
    }
#else
    for (int i = tid; i < 512; i += 256) {            // 512 chunks of 8 bf16
      const int row = i >> 2, c8 = (i & 3) * 8;
      *(v8bf*)&As[ib * 4096 + row * 32 + c8] =
          *(const v8bf*)&A[(size_t)(m0 + row) * K + kk + c8];
      *(v8bf*)&Ws[ib * 4096 + row * 32 + c8] =
          *(const v8bf*)&W[(size_t)(n0 + row) * K + kk + c8];
    }
#endif
    __syncthreads();

    const bf16* Ab = &As[ib * 4096];
    const bf16* Wb = &Ws[ib * 4096];

    // A fragment: lane covers row (wave*16+r), K = hh*8+{0..7} and 16+hh*8+{0..7}
    const int arow = wave * 16 + r;
    v8bf alo = *(const v8bf*)&Ab[arow * 32 + hh * 8];
    v8bf ahi = *(const v8bf*)&Ab[arow * 32 + 16 + hh * 8];
    v16bf a;
#pragma unroll
    for (int i = 0; i < 8; ++i) { a[i] = alo[i]; a[i + 8] = ahi[i]; }

    // B fragments: lane covers col (s*16+r), K = hh*8+{0..7} and 16+hh*8+{0..7}.
    // Rotating 2-deep pipeline: loads for s+1 issue before WMMA s.
    const bf16* Wrow = &Wb[r * 32 + hh * 8];
    v8bf blo = *(const v8bf*)&Wrow[0];
    v8bf bhi = *(const v8bf*)&Wrow[16];
#pragma unroll
    for (int s = 0; s < 8; ++s) {
      v16bf b;
#pragma unroll
      for (int i = 0; i < 8; ++i) { b[i] = blo[i]; b[i + 8] = bhi[i]; }
      if (s < 7) {
        blo = *(const v8bf*)&Wrow[(s + 1) * 16 * 32];
        bhi = *(const v8bf*)&Wrow[(s + 1) * 16 * 32 + 16];
      }
      acc[s] = __builtin_amdgcn_wmma_f32_16x16x32_bf16(
          false, a, false, b, (short)0, acc[s], false, false);
    }
  }

  // Epilogue: C/D layout -> VGPR v holds row hh*8+v, col = r.
  const int mbase = m0 + wave * 16 + hh * 8;
#pragma unroll
  for (int s = 0; s < 8; ++s) {
    const int n = n0 + s * 16 + r;
    const float bsum = bia[n] + bib[n];
#pragma unroll
    for (int v = 0; v < 8; ++v)
      C[(size_t)(mbase + v) * N + n] = acc[s][v] + bsum;
  }
}

// ---------------------------------------------------------------------------
// Layer-1 scan: 64 persistent WGs x 256 threads. WG owns 16 hidden units
// (weight-stationary in L2, bf16). Thread (u=tid>>4, s=tid&15): 64-wide dot
// segment per gate, LDS tree-reduce, double-buffered h, grid barrier/step.
// ---------------------------------------------------------------------------
__global__ void __launch_bounds__(256) lstm1_scan(
    const float* __restrict__ xg, const bf16* __restrict__ Whh,
    bf16* __restrict__ hsB, float* __restrict__ hbuf,
    unsigned int* bar, int T)
{
  __shared__ float hsh[H1];
  __shared__ float red[16][17];
  __shared__ float gates[16][4];
  __shared__ float cst[16];
  const int tid = threadIdx.x;
  const int nb  = blockIdx.x * 16;
  const int u   = tid >> 4, sgm = tid & 15;
  if (tid < 16) {
    cst[tid] = 0.0f;
    hbuf[nb + tid] = 0.0f;
    hbuf[H1 + nb + tid] = 0.0f;
  }
  unsigned int bc = 0;
  gsync(bar, &bc);
  for (int t = 0; t < T; ++t) {
    const int cur = t & 1, nxt = cur ^ 1;
    for (int i = tid; i < H1; i += 256) hsh[i] = hbuf[cur * H1 + i];
    __syncthreads();
#pragma unroll
    for (int g = 0; g < 4; ++g) {
      const int row = g * H1 + nb + u;
      const v8bf* wv = (const v8bf*)(Whh + (size_t)row * H1 + sgm * 64);
      const float* hv = &hsh[sgm * 64];
      float p = 0.0f;
#pragma unroll
      for (int c = 0; c < 8; ++c) {
        v8bf wb = wv[c];
#pragma unroll
        for (int j = 0; j < 8; ++j) p += b2f(wb[j]) * hv[c * 8 + j];
      }
      red[u][sgm] = p;
      __syncthreads();
      if (sgm == 0) {
        float a = 0.0f;
#pragma unroll
        for (int j = 0; j < 16; ++j) a += red[u][j];
        gates[u][g] = a + xg[(size_t)t * G1 + row];
      }
      __syncthreads();
    }
    if (tid < 16) {
      const float i_ = sigf(gates[tid][0]);
      const float f_ = sigf(gates[tid][1]);
      const float g_ = tanhf(gates[tid][2]);
      const float o_ = sigf(gates[tid][3]);
      const float c  = f_ * cst[tid] + i_ * g_;
      cst[tid] = c;
      const float h = o_ * tanhf(c);
      hbuf[nxt * H1 + nb + tid] = h;
      hsB[(size_t)t * H1 + nb + tid] = f2b(h);   // bf16 feed for GEMM2
    }
    gsync(bar, &bc);
  }
}

// ---------------------------------------------------------------------------
// Layer-2 scan: WG owns 8 units; thread (u=tid>>5, s=tid&31): 16-wide segment.
// Only final h is needed -> written to d_out at t==T-1.
// ---------------------------------------------------------------------------
__global__ void __launch_bounds__(256) lstm2_scan(
    const float* __restrict__ xg, const bf16* __restrict__ Whh,
    float* __restrict__ hbuf, float* __restrict__ out,
    unsigned int* bar, int T)
{
  __shared__ float hsh[H2];
  __shared__ float red[8][33];
  __shared__ float gates[8][4];
  __shared__ float cst[8];
  const int tid = threadIdx.x;
  const int nb  = blockIdx.x * 8;
  const int u   = tid >> 5, sgm = tid & 31;
  if (tid < 8) {
    cst[tid] = 0.0f;
    hbuf[nb + tid] = 0.0f;
    hbuf[H2 + nb + tid] = 0.0f;
  }
  unsigned int bc = 0;
  gsync(bar, &bc);
  for (int t = 0; t < T; ++t) {
    const int cur = t & 1, nxt = cur ^ 1;
    for (int i = tid; i < H2; i += 256) hsh[i] = hbuf[cur * H2 + i];
    __syncthreads();
#pragma unroll
    for (int g = 0; g < 4; ++g) {
      const int row = g * H2 + nb + u;
      const v8bf* wv = (const v8bf*)(Whh + (size_t)row * H2 + sgm * 16);
      const float* hv = &hsh[sgm * 16];
      float p = 0.0f;
#pragma unroll
      for (int c = 0; c < 2; ++c) {
        v8bf wb = wv[c];
#pragma unroll
        for (int j = 0; j < 8; ++j) p += b2f(wb[j]) * hv[c * 8 + j];
      }
      red[u][sgm] = p;
      __syncthreads();
      if (sgm == 0) {
        float a = 0.0f;
#pragma unroll
        for (int j = 0; j < 32; ++j) a += red[u][j];
        gates[u][g] = a + xg[(size_t)t * G2 + row];
      }
      __syncthreads();
    }
    if (tid < 8) {
      const float i_ = sigf(gates[tid][0]);
      const float f_ = sigf(gates[tid][1]);
      const float g_ = tanhf(gates[tid][2]);
      const float o_ = sigf(gates[tid][3]);
      const float c  = f_ * cst[tid] + i_ * g_;
      cst[tid] = c;
      const float h = o_ * tanhf(c);
      hbuf[nxt * H2 + nb + tid] = h;
      if (t == T - 1) out[nb + tid] = h;
    }
    gsync(bar, &bc);
  }
}

// ---------------------------------------------------------------------------
// Utility kernels
// ---------------------------------------------------------------------------
__global__ void cvt_f32_bf16(const float* __restrict__ s, bf16* __restrict__ d, size_t n) {
  size_t i  = (size_t)blockIdx.x * blockDim.x + threadIdx.x;
  size_t st = (size_t)gridDim.x * blockDim.x;
  for (; i < n; i += st) d[i] = f2b(s[i]);
}
__global__ void zero_u32(unsigned int* p, int n) {
  if ((int)threadIdx.x < n) p[threadIdx.x] = 0u;
}

// ---------------------------------------------------------------------------
// Host launch
// ---------------------------------------------------------------------------
extern "C" void kernel_launch(void* const* d_in, const int* in_sizes, int n_in,
                              void* d_out, int out_size, void* d_ws, size_t ws_size,
                              hipStream_t stream)
{
  (void)in_sizes; (void)n_in; (void)out_size; (void)ws_size;
  const float* x    = (const float*)d_in[0];
  const float* Wih1 = (const float*)d_in[1];
  const float* Whh1 = (const float*)d_in[2];
  const float* bih1 = (const float*)d_in[3];
  const float* bhh1 = (const float*)d_in[4];
  const float* Wih2 = (const float*)d_in[5];
  const float* Whh2 = (const float*)d_in[6];
  const float* bih2 = (const float*)d_in[7];
  const float* bhh2 = (const float*)d_in[8];
  float* out = (float*)d_out;

  // Workspace carve-out (~470 MB total)
  char* ws = (char*)d_ws;
  size_t off = 0;
  auto carve = [&](size_t bytes) -> char* {
    char* p = ws + off;
    off = (off + bytes + 255) & ~(size_t)255;
    return p;
  };
  unsigned int* bar = (unsigned int*)carve(256);
  bf16*  xB    = (bf16*)carve((size_t)T_LEN * FEAT * 2);
  bf16*  Wih1B = (bf16*)carve((size_t)G1 * FEAT * 2);
  bf16*  Wih2B = (bf16*)carve((size_t)G2 * H1 * 2);
  bf16*  Whh1B = (bf16*)carve((size_t)G1 * H1 * 2);
  bf16*  Whh2B = (bf16*)carve((size_t)G2 * H2 * 2);
  float* xg1   = (float*)carve((size_t)T_LEN * G1 * 4);
  float* xg2   = (float*)carve((size_t)T_LEN * G2 * 4);
  bf16*  hs1B  = (bf16*)carve((size_t)T_LEN * H1 * 2);
  float* h1b   = (float*)carve((size_t)2 * H1 * 4);
  float* h2b   = (float*)carve((size_t)2 * H2 * 4);

  zero_u32<<<1, 32, 0, stream>>>(bar, 2);

  cvt_f32_bf16<<<512, 256, 0, stream>>>(x,    xB,    (size_t)T_LEN * FEAT);
  cvt_f32_bf16<<<256, 256, 0, stream>>>(Wih1, Wih1B, (size_t)G1 * FEAT);
  cvt_f32_bf16<<<256, 256, 0, stream>>>(Wih2, Wih2B, (size_t)G2 * H1);
  cvt_f32_bf16<<<256, 256, 0, stream>>>(Whh1, Whh1B, (size_t)G1 * H1);
  cvt_f32_bf16<<<256, 256, 0, stream>>>(Whh2, Whh2B, (size_t)G2 * H2);

  // xg1 = x @ W_ih1^T + (b_ih1 + b_hh1)
  gemm_bf16_wmma<<<dim3(G1 / 128, T_LEN / 128), 256, 0, stream>>>(
      xB, Wih1B, bih1, bhh1, xg1, G1, FEAT);

  // layer-1 recurrence -> hs1 (bf16)
  lstm1_scan<<<NWG, 256, 0, stream>>>(xg1, Whh1B, hs1B, h1b, bar + 0, T_LEN);

  // xg2 = hs1 @ W_ih2^T + (b_ih2 + b_hh2)
  gemm_bf16_wmma<<<dim3(G2 / 128, T_LEN / 128), 256, 0, stream>>>(
      hs1B, Wih2B, bih2, bhh2, xg2, G2, H1);

  // layer-2 recurrence -> final h to d_out
  lstm2_scan<<<NWG, 256, 0, stream>>>(xg2, Whh2B, h2b, out, bar + 1, T_LEN);
}
